// Quantizer_57896159150132
// MI455X (gfx1250) — matrix operand order repaired and checked by reading, not compiled, not run on hardware
//
#include <hip/hip_runtime.h>
#include <hip/hip_bf16.h>
#include <math.h>

// ---------------------------------------------------------------------------
// VQ-VAE quantizer for MI455X (gfx1250, wave32, WMMA).
//
// Outputs (flat, f32): [0] loss | [1..8388609) z_q | [8388609] perplexity |
//                      [8388610..41943042) min_encodings | [..+32768) idx
// ---------------------------------------------------------------------------

typedef __attribute__((ext_vector_type(16))) _Float16 v16h;
typedef __attribute__((ext_vector_type(4)))  _Float16 v4h;
typedef __attribute__((ext_vector_type(8)))  float    v8f;

#define N_PIX   32768      // 32*32*32 pixel vectors
#define E_DIM   256
#define N_E     1024
#define AS      264        // padded LDS stride in halves (bank-conflict free)

// ws layout (bytes): [0,4096) int counts[1024]; [4096,8192) float norms[1024];
//                    [8192,8196) float lossAcc; [8704, 8704+512K) f16 emb copy
#define WS_COUNTS(ws)  ((int*)(ws))
#define WS_NORMS(ws)   ((float*)((char*)(ws) + 4096))
#define WS_LOSS(ws)    ((float*)((char*)(ws) + 8192))
#define WS_EMB16(ws)   ((_Float16*)((char*)(ws) + 8704))
#define WS_NEED_F16    (8704 + (size_t)N_E * E_DIM * 2)

#define ZQ_OFF   ((size_t)1)
#define PERP_OFF ((size_t)8388609)
#define ME_OFF   ((size_t)8388610)
#define IDX_OFF  ((size_t)41943042)

__global__ void vq_init(int* counts, float* lossAcc) {
  int t = threadIdx.x;
  for (int j = t; j < N_E; j += 256) counts[j] = 0;
  if (t == 0) *lossAcc = 0.0f;
}

// One wave per codebook row: norms[j] = sum_k emb[j][k]^2
__global__ void vq_emb_norms(const float* __restrict__ emb,
                             float* __restrict__ norms) {
  int gw   = (blockIdx.x * blockDim.x + threadIdx.x) >> 5;
  int lane = threadIdx.x & 31;
  if (gw >= N_E) return;
  const float* r = emb + (size_t)gw * E_DIM;
  float s = 0.0f;
  for (int i = lane; i < E_DIM; i += 32) { float v = r[i]; s += v * v; }
  #pragma unroll
  for (int m = 16; m; m >>= 1) s += __shfl_xor(s, m, 32);
  if (lane == 0) norms[gw] = s;
}

// One-time f32 -> f16 codebook conversion (halves hot-loop L2 bytes, kills cvts)
__global__ void vq_emb_f16(const float* __restrict__ emb,
                           _Float16* __restrict__ emb16) {
  int i = (blockIdx.x * 256 + threadIdx.x) * 4;   // 256 blocks cover 262144
  float4 f = *(const float4*)(emb + i);
  v4h o = {(_Float16)f.x, (_Float16)f.y, (_Float16)f.z, (_Float16)f.w};
  *(v4h*)(emb16 + i) = o;
}

__device__ __forceinline__ v16h bfrag_from_f32(const float* bp) {
  float4 f0 = ((const float4*)bp)[0];
  float4 f1 = ((const float4*)bp)[1];
  float4 f2 = ((const float4*)bp)[2];
  float4 f3 = ((const float4*)bp)[3];
  v16h b;
  b[0]=(_Float16)f0.x;  b[1]=(_Float16)f0.y;  b[2]=(_Float16)f0.z;  b[3]=(_Float16)f0.w;
  b[4]=(_Float16)f1.x;  b[5]=(_Float16)f1.y;  b[6]=(_Float16)f1.z;  b[7]=(_Float16)f1.w;
  b[8]=(_Float16)f2.x;  b[9]=(_Float16)f2.y;  b[10]=(_Float16)f2.z; b[11]=(_Float16)f2.w;
  b[12]=(_Float16)f3.x; b[13]=(_Float16)f3.y; b[14]=(_Float16)f3.z; b[15]=(_Float16)f3.w;
  return b;
}

// Main: per 16-pixel tile, f16 WMMA distance matmul vs all 1024 codes,
// fused argmin + all outputs. PRE16: codebook prestaged as f16 in ws.
template <bool PRE16>
__global__ void __launch_bounds__(256)
vq_main(const float* __restrict__ z,
        const float* __restrict__ emb,
        const _Float16* __restrict__ emb16,
        const float* __restrict__ norms,
        float* __restrict__ zq,
        float* __restrict__ me,
        float* __restrict__ idxOut,
        int*   __restrict__ counts,
        float* __restrict__ lossAcc) {
  __shared__ _Float16 zA[16 * AS];        // f16 z tile, [pixel][channel]
  __shared__ float    candS[8][16];
  __shared__ int      candI[8][16];
  __shared__ int      finalIdx[16];

  const int tid  = threadIdx.x;
  const int lane = tid & 31;
  const int wave = tid >> 5;
  const int n0   = blockIdx.x * 16;       // first pixel of tile
  const int b    = n0 >> 10;              // batch index
  const int hw0  = n0 & 1023;             // h*32+w base (tile stays in one row)
  const size_t zBase = (size_t)b * 262144 + (size_t)hw0;

  // ---- Stage z tile into LDS as f16 (coalesced: 16 consecutive pixels) ----
  {
    int p = tid & 15, c0 = tid >> 4;
    for (int c = c0; c < E_DIM; c += 16)
      zA[p * AS + c] = (_Float16)z[zBase + (size_t)c * 1024 + p];
  }
  __syncthreads();

  // ---- Preload A fragments, contiguous k order k = k0 + 16*hi + s (16x32 per k-step) ----
  const int row = lane & 15;
  const int hi  = lane >> 4;
  v16h aF[8];
  #pragma unroll
  for (int ks = 0; ks < 8; ++ks) {
    const int k0 = ks * 32;
    #pragma unroll
    for (int e = 0; e < 8; ++e) {
      aF[ks][e]     = zA[row * AS + k0 + hi * 16 + e];
      aF[ks][8 + e] = zA[row * AS + k0 + hi * 16 + 8 + e];
    }
  }

  // ---- 8 col-tiles of 16 codes per wave, processed 2 at a time so the two
  //      WMMA accumulator chains interleave (hides WMMA RAW latency) --------
  const int nidx = lane & 15;             // N (code) column owned by this lane
  float bestS[8];
  int   bestI[8];
  #pragma unroll
  for (int v = 0; v < 8; ++v) { bestS[v] = 3.0e38f; bestI[v] = 0; }

  for (int jt = 0; jt < 8; jt += 2) {
    const int j0 = (wave * 8 + jt) * 16;
    const int j1 = j0 + 16;
    v8f acc0 = {};
    v8f acc1 = {};
    #pragma unroll
    for (int ks = 0; ks < 8; ++ks) {
      // B fragment (32x16): lanes 0-15 hold K=k0..k0+15 of col j0+nidx,
      // lanes 16-31 hold K=k0+16..k0+31 (16 consecutive elems per lane).
      const size_t koff = (size_t)ks * 32 + (size_t)hi * 16;
      v16h b0, b1;
      if (PRE16) {
        b0 = *(const v16h*)(emb16 + (size_t)(j0 + nidx) * E_DIM + koff);
        b1 = *(const v16h*)(emb16 + (size_t)(j1 + nidx) * E_DIM + koff);
      } else {
        b0 = bfrag_from_f32(emb + (size_t)(j0 + nidx) * E_DIM + koff);
        b1 = bfrag_from_f32(emb + (size_t)(j1 + nidx) * E_DIM + koff);
      }
      acc0 = __builtin_amdgcn_wmma_f32_16x16x32_f16(
          false, aF[ks], false, b0, (short)0, acc0, false, false);
      acc1 = __builtin_amdgcn_wmma_f32_16x16x32_f16(
          false, aF[ks], false, b1, (short)0, acc1, false, false);
    }
    // score = ||e_j||^2 - 2 * (z . e_j)   (||z||^2 constant per pixel)
    const float nrm0 = norms[j0 + nidx];
    const float nrm1 = norms[j1 + nidx];
    #pragma unroll
    for (int v = 0; v < 8; ++v) {
      float s0 = fmaf(-2.0f, acc0[v], nrm0);
      if (s0 < bestS[v]) { bestS[v] = s0; bestI[v] = j0 + nidx; }
      float s1 = fmaf(-2.0f, acc1[v], nrm1);
      if (s1 < bestS[v]) { bestS[v] = s1; bestI[v] = j1 + nidx; }
    }
  }

  // ---- Reduce argmin across the 16 lanes of each half (N dimension) ----
  #pragma unroll
  for (int v = 0; v < 8; ++v) {
    #pragma unroll
    for (int m = 1; m < 16; m <<= 1) {
      float os = __shfl_xor(bestS[v], m, 32);
      int   oi = __shfl_xor(bestI[v], m, 32);
      if (os < bestS[v] || (os == bestS[v] && oi < bestI[v])) {
        bestS[v] = os; bestI[v] = oi;
      }
    }
    if (nidx == 0) {                      // lanes 0 and 16: pixels v, v+8
      candS[wave][v + hi * 8] = bestS[v];
      candI[wave][v + hi * 8] = bestI[v];
    }
  }
  __syncthreads();

  // ---- Cross-wave reduction; idx / histogram outputs ----
  if (tid < 16) {
    float bs = candS[0][tid];
    int   bi = candI[0][tid];
    #pragma unroll
    for (int w = 1; w < 8; ++w) {
      float s = candS[w][tid]; int i = candI[w][tid];
      if (s < bs || (s == bs && i < bi)) { bs = s; bi = i; }
    }
    finalIdx[tid] = bi;
    idxOut[n0 + tid] = (float)bi;
    atomicAdd(&counts[bi], 1);
  }
  __syncthreads();

  // ---- z_q (straight-through == emb[idx]) back to NCHW, + loss partial ----
  {
    int p = tid & 15, c0 = tid >> 4;
    const int fi = finalIdx[p];
    const float* erow = emb + (size_t)fi * E_DIM;
    float lsum = 0.0f;
    const size_t ob = zBase + p;
    for (int c = c0; c < E_DIM; c += 16) {
      float e = erow[c];
      zq[ob + (size_t)c * 1024] = e;
      float d = e - (float)zA[p * AS + c];
      lsum = fmaf(d, d, lsum);
    }
    #pragma unroll
    for (int m = 16; m; m >>= 1) lsum += __shfl_xor(lsum, m, 32);
    if (lane == 0) atomicAdd(lossAcc, lsum);
  }

  // ---- Full one-hot rows (single pass, coalesced 64B chunks) ----
  {
    int p = tid >> 4, q = tid & 15;
    const int fi = finalIdx[p];
    float* mrow = me + (size_t)(n0 + p) * N_E;
    for (int j = q; j < N_E; j += 16) mrow[j] = (j == fi) ? 1.0f : 0.0f;
  }
}

__global__ void vq_finalize(const int* __restrict__ counts,
                            const float* __restrict__ lossAcc,
                            float* __restrict__ outLoss,
                            float* __restrict__ outPerp) {
  __shared__ float red[256];
  int t = threadIdx.x;
  float h = 0.0f;
  for (int j = t; j < N_E; j += 256) {
    float em = (float)counts[j] * (1.0f / (float)N_PIX);
    h += em * logf(em + 1e-10f);
  }
  red[t] = h;
  __syncthreads();
  for (int s = 128; s; s >>= 1) {
    if (t < s) red[t] += red[t + s];
    __syncthreads();
  }
  if (t == 0) {
    *outPerp = expf(-red[0]);
    // loss = mean((zq-z)^2) + 0.25*mean((zq-z)^2); mean over 32*256*32*32 elems
    *outLoss = 1.25f * (*lossAcc) * (1.0f / 8388608.0f);
  }
}

extern "C" void kernel_launch(void* const* d_in, const int* in_sizes, int n_in,
                              void* d_out, int out_size, void* d_ws, size_t ws_size,
                              hipStream_t stream) {
  const float* z   = (const float*)d_in[0];   // (32,256,32,32) f32
  const float* emb = (const float*)d_in[1];   // (1024,256) f32
  float* out = (float*)d_out;

  int*      counts  = WS_COUNTS(d_ws);
  float*    norms   = WS_NORMS(d_ws);
  float*    lossAcc = WS_LOSS(d_ws);
  _Float16* emb16   = WS_EMB16(d_ws);

  vq_init<<<1, 256, 0, stream>>>(counts, lossAcc);
  vq_emb_norms<<<128, 256, 0, stream>>>(emb, norms);

  if (ws_size >= WS_NEED_F16) {
    vq_emb_f16<<<256, 256, 0, stream>>>(emb, emb16);
    vq_main<true><<<N_PIX / 16, 256, 0, stream>>>(
        z, emb, emb16, norms,
        out + ZQ_OFF, out + ME_OFF, out + IDX_OFF, counts, lossAcc);
  } else {
    vq_main<false><<<N_PIX / 16, 256, 0, stream>>>(
        z, emb, emb16, norms,
        out + ZQ_OFF, out + ME_OFF, out + IDX_OFF, counts, lossAcc);
  }
  vq_finalize<<<1, 256, 0, stream>>>(counts, lossAcc, out, out + PERP_OFF);
}
